// ThreeConv_57157424775210
// MI455X (gfx1250) — compile-verified, matrix operand order checked by reading, not run on hardware
//
#include <hip/hip_runtime.h>
#include <math.h>

typedef __attribute__((ext_vector_type(16))) _Float16 v16h;
typedef __attribute__((ext_vector_type(8)))  float    v8f;
typedef __attribute__((ext_vector_type(4)))  unsigned int u32x4;
typedef __attribute__((ext_vector_type(8)))  int      i32x8;
typedef __attribute__((ext_vector_type(4)))  int      i32x4;

#define HEADS 4

// ---------------------------------------------------------------------------
// util: fill a float buffer
// ---------------------------------------------------------------------------
__global__ void fill_f32(float* __restrict__ p, int n, float v) {
  int i = blockIdx.x * blockDim.x + threadIdx.x;
  if (i < n) p[i] = v;
}

// ---------------------------------------------------------------------------
// degree: deg[dst] += 1 over all edges (shared by all three conv layers)
// ---------------------------------------------------------------------------
__global__ void degree_kernel(const int* __restrict__ dst, int E, float* __restrict__ deg) {
  int i = blockIdx.x * blockDim.x + threadIdx.x;
  if (i < E) atomicAdd(&deg[dst[i]], 1.0f);
}

// ---------------------------------------------------------------------------
// T = X @ U   (node-level attention projection, [N, HEADS], K in {16,32})
// ---------------------------------------------------------------------------
__global__ void proj_u_kernel(const float* __restrict__ X, const float* __restrict__ U,
                              float* __restrict__ T, int N, int K) {
  int n = blockIdx.x * blockDim.x + threadIdx.x;
  if (n >= N) return;
  float acc0 = 0.f, acc1 = 0.f, acc2 = 0.f, acc3 = 0.f;
  const float* xr = X + n * K;
  for (int k = 0; k < K; ++k) {
    float xv = xr[k];
    acc0 += xv * U[k * HEADS + 0];
    acc1 += xv * U[k * HEADS + 1];
    acc2 += xv * U[k * HEADS + 2];
    acc3 += xv * U[k * HEADS + 3];
  }
  T[n * HEADS + 0] = acc0;
  T[n * HEADS + 1] = acc1;
  T[n * HEADS + 2] = acc2;
  T[n * HEADS + 3] = acc3;
}

// ---------------------------------------------------------------------------
// WMMA GEMM: Y[N,C] = X[N,K] @ W[K,C]   (K <= 32, C % 32 == 0)
// One wave computes a 16x32 output tile = 2x v_wmma_f32_16x16x32_f16 sharing
// the packed A operand. 8 waves/block => 128 rows per block; grid.y = C/32.
// W tile (K x 32 f32, row stride C) is staged LDS via the Tensor Data Mover
// (tensor_load_to_lds + s_wait_tensorcnt); X tiles staged per-wave as f16.
// ---------------------------------------------------------------------------
__global__ void gemm_wmma_kernel(const float* __restrict__ X, const float* __restrict__ W,
                                 float* __restrict__ Y, int N, int K, int C) {
  __shared__ float    sW[32 * 32];      // K-major W tile (f32, TDM-staged)
  __shared__ _Float16 sX[8][16 * 32];   // per-wave 16x32 row-major X tile (f16)

  const int tid = threadIdx.x;
  const int wave = tid >> 5;
  const int lane = tid & 31;
  const int colBase = blockIdx.y * 32;
  const int rowBase = blockIdx.x * 128 + wave * 16;

  // Zero-pad W rows K..31 (needed when K==16); disjoint from TDM target region.
  for (int idx = tid + K * 32; idx < 32 * 32; idx += 256) sW[idx] = 0.0f;

  if (wave == 0) {
    // Tensor DMA descriptor: 2D tile, element=4B, tile 32 x K, row stride C.
    unsigned lds = (unsigned)(uintptr_t)(&sW[0]);          // flat addr[31:0] == LDS offset
    unsigned long long ga = (unsigned long long)(uintptr_t)(W + colBase);
    u32x4 g0;
    g0[0] = 1u;                                            // count=1, user descriptor
    g0[1] = lds;                                           // lds_addr (bytes)
    g0[2] = (unsigned)(ga & 0xFFFFFFFFu);                  // global_addr[31:0]
    g0[3] = (unsigned)((ga >> 32) & 0x01FFFFFFu) | (2u << 30); // addr[56:32] | type=2
    i32x8 g1;
    g1[0] = (2 << 16);                                     // data_size=2 (4 bytes)
    g1[1] = (C & 0xFFFF) << 16;                            // tensor_dim0[15:0]
    g1[2] = (K & 0xFFFF) << 16;                            // dim0 hi=0 | tensor_dim1[15:0]
    g1[3] = (32 << 16);                                    // dim1 hi=0 | tile_dim0=32
    g1[4] = K;                                             // tile_dim1=K, tile_dim2=0
    g1[5] = C;                                             // tensor_dim0_stride[31:0]
    g1[6] = 0;
    g1[7] = 0;
    i32x4 gz = {0, 0, 0, 0};
#if __clang_major__ >= 23
    i32x8 gz8 = {0, 0, 0, 0, 0, 0, 0, 0};
    __builtin_amdgcn_tensor_load_to_lds(g0, g1, gz, gz, gz8, 0);
#else
    __builtin_amdgcn_tensor_load_to_lds(g0, g1, gz, gz, 0);
#endif
    __builtin_amdgcn_s_wait_tensorcnt(0);
  }

  // Stage this wave's 16x32 X tile (f32 -> f16)
  for (int idx = lane; idx < 16 * 32; idx += 32) {
    int r = idx >> 5;
    int k = idx & 31;
    int node = rowBase + r;
    float v = (node < N && k < K) ? X[node * K + k] : 0.0f;
    sX[wave][idx] = (_Float16)v;
  }
  __syncthreads();

  // Pack A (16x32 f16): lanes 0-15 -> M=lane,   K=0..7  & 16..23
  //                     lanes 16-31 -> M=lane-16, K=8..15 & 24..31
  const int halfsel = lane >> 4;
  const int kb = halfsel * 8;
  const int rm = lane & 15;

  v16h a, b0, b1;
#pragma unroll
  for (int j = 0; j < 8; ++j) {
    a[j]      = sX[wave][rm * 32 + kb + j];
    a[j + 8]  = sX[wave][rm * 32 + 16 + kb + j];
    // B (32x16 f16): column N = lane&15, same K striping as A
    b0[j]     = (_Float16)sW[(kb + j) * 32 + rm];
    b0[j + 8] = (_Float16)sW[(16 + kb + j) * 32 + rm];
    b1[j]     = (_Float16)sW[(kb + j) * 32 + 16 + rm];
    b1[j + 8] = (_Float16)sW[(16 + kb + j) * 32 + 16 + rm];
  }

  v8f acc0 = {}, acc1 = {};
  acc0 = __builtin_amdgcn_wmma_f32_16x16x32_f16(false, a, false, b0, (short)0, acc0, false, false);
  acc1 = __builtin_amdgcn_wmma_f32_16x16x32_f16(false, a, false, b1, (short)0, acc1, false, false);

  // C/D layout: VGPR r, lanes 0-15 -> M=r, lanes 16-31 -> M=r+8, N=lane&15
  const int col0 = colBase + rm;
  const int mrow = rowBase + halfsel * 8;
  if (rowBase + 16 <= N) {
    int base = mrow * C + col0;
#pragma unroll
    for (int r = 0; r < 8; ++r) {
      Y[base + r * C]      = acc0[r];
      Y[base + r * C + 16] = acc1[r];
    }
  } else {
#pragma unroll
    for (int r = 0; r < 8; ++r) {
      int node = mrow + r;
      if (node < N) {
        Y[node * C + col0]      = acc0[r];
        Y[node * C + col0 + 16] = acc1[r];
      }
    }
  }
}

// ---------------------------------------------------------------------------
// Edge phase: wave-per-edge. Softmax over 4 heads of (T[src]-T[dst]+c),
// lanes sweep Fout, gather Y[src] (L2-resident), scatter-atomic into agg[dst].
// ---------------------------------------------------------------------------
__global__ void edge_kernel(const int* __restrict__ ei, int E,
                            const float* __restrict__ T, const float* __restrict__ cvec,
                            const float* __restrict__ Y, float* __restrict__ agg,
                            int Fout) {
  const int wave = threadIdx.x >> 5;
  const int lane = threadIdx.x & 31;
  const int edge = blockIdx.x * 8 + wave;
  if (edge >= E) return;

  const int src = ei[edge];
  const int dst = ei[E + edge];

  float s0 = T[src * HEADS + 0] - T[dst * HEADS + 0] + cvec[0];
  float s1 = T[src * HEADS + 1] - T[dst * HEADS + 1] + cvec[1];
  float s2 = T[src * HEADS + 2] - T[dst * HEADS + 2] + cvec[2];
  float s3 = T[src * HEADS + 3] - T[dst * HEADS + 3] + cvec[3];

  const int C = HEADS * Fout;
  const float* ys = Y + src * C;
  float* ag = agg + dst * Fout;
  __builtin_prefetch(ys + lane, 0, 1);   // global_prefetch_b8: warm Y row

  float m = fmaxf(fmaxf(s0, s1), fmaxf(s2, s3));
  float q0 = __expf(s0 - m), q1 = __expf(s1 - m), q2 = __expf(s2 - m), q3 = __expf(s3 - m);
  float inv = 1.0f / (q0 + q1 + q2 + q3);
  q0 *= inv; q1 *= inv; q2 *= inv; q3 *= inv;

  for (int o = lane; o < Fout; o += 32) {
    float acc = q0 * ys[o] + q1 * ys[Fout + o] + q2 * ys[2 * Fout + o] + q3 * ys[3 * Fout + o];
    atomicAdd(&ag[o], acc);
  }
}

// ---------------------------------------------------------------------------
// Finalize: H = relu(agg / max(deg,1) + b)
// ---------------------------------------------------------------------------
__global__ void finalize_kernel(const float* __restrict__ agg, const float* __restrict__ deg,
                                const float* __restrict__ bias, float* __restrict__ H,
                                int N, int Fout) {
  int i = blockIdx.x * blockDim.x + threadIdx.x;
  if (i >= N * Fout) return;
  int n = i / Fout;
  int f = i - n * Fout;
  float d = fmaxf(deg[n], 1.0f);
  H[i] = fmaxf(agg[i] / d + bias[f], 0.0f);
}

// ---------------------------------------------------------------------------
// MLP tail: 64 -> 32 -> 16 -> 8 -> 4 -> 1, ReLU between, sigmoid at end.
// Weights staged in LDS once per block; one node per thread.
// ---------------------------------------------------------------------------
__global__ void mlp_tail_kernel(const float* __restrict__ H,
                                const float* __restrict__ lw1, const float* __restrict__ lb1,
                                const float* __restrict__ lw2, const float* __restrict__ lb2,
                                const float* __restrict__ lw3, const float* __restrict__ lb3,
                                const float* __restrict__ lw4, const float* __restrict__ lb4,
                                const float* __restrict__ lw5, const float* __restrict__ lb5,
                                float* __restrict__ out, int N) {
  __shared__ float w1[64 * 32], w2[32 * 16], w3[16 * 8], w4[8 * 4], w5[4];
  __shared__ float b1[32], b2[16], b3[8], b4[4], b5[1];
  const int tid = threadIdx.x;
  for (int i = tid; i < 64 * 32; i += blockDim.x) w1[i] = lw1[i];
  for (int i = tid; i < 32 * 16; i += blockDim.x) w2[i] = lw2[i];
  for (int i = tid; i < 16 * 8;  i += blockDim.x) w3[i] = lw3[i];
  for (int i = tid; i < 8 * 4;   i += blockDim.x) w4[i] = lw4[i];
  if (tid < 32) b1[tid] = lb1[tid];
  if (tid < 16) b2[tid] = lb2[tid];
  if (tid < 8)  b3[tid] = lb3[tid];
  if (tid < 4)  { b4[tid] = lb4[tid]; w5[tid] = lw5[tid]; }
  if (tid == 0) b5[0] = lb5[0];
  __syncthreads();

  int n = blockIdx.x * blockDim.x + tid;
  if (n >= N) return;

  float h0[64];
#pragma unroll
  for (int k = 0; k < 64; ++k) h0[k] = H[n * 64 + k];

  float h1[32];
#pragma unroll
  for (int j = 0; j < 32; ++j) {
    float a = b1[j];
    for (int k = 0; k < 64; ++k) a += h0[k] * w1[k * 32 + j];
    h1[j] = fmaxf(a, 0.0f);
  }
  float h2[16];
#pragma unroll
  for (int j = 0; j < 16; ++j) {
    float a = b2[j];
    for (int k = 0; k < 32; ++k) a += h1[k] * w2[k * 16 + j];
    h2[j] = fmaxf(a, 0.0f);
  }
  float h3[8];
#pragma unroll
  for (int j = 0; j < 8; ++j) {
    float a = b3[j];
    for (int k = 0; k < 16; ++k) a += h2[k] * w3[k * 8 + j];
    h3[j] = fmaxf(a, 0.0f);
  }
  float h4[4];
#pragma unroll
  for (int j = 0; j < 4; ++j) {
    float a = b4[j];
    for (int k = 0; k < 8; ++k) a += h3[k] * w4[k * 4 + j];
    h4[j] = fmaxf(a, 0.0f);
  }
  float z = b5[0];
#pragma unroll
  for (int k = 0; k < 4; ++k) z += h4[k] * w5[k];
  out[n] = 1.0f / (1.0f + __expf(-z));
}

// ---------------------------------------------------------------------------
// Driver
// ---------------------------------------------------------------------------
extern "C" void kernel_launch(void* const* d_in, const int* in_sizes, int n_in,
                              void* d_out, int out_size, void* d_ws, size_t ws_size,
                              hipStream_t stream) {
  const float* x  = (const float*)d_in[0];
  const int*   ei = (const int*)d_in[1];

  const float* Uw[3]  = {(const float*)d_in[2],  (const float*)d_in[6],  (const float*)d_in[10]};
  const float* cv[3]  = {(const float*)d_in[3],  (const float*)d_in[7],  (const float*)d_in[11]};
  const float* Ww[3]  = {(const float*)d_in[4],  (const float*)d_in[8],  (const float*)d_in[12]};
  const float* bw[3]  = {(const float*)d_in[5],  (const float*)d_in[9],  (const float*)d_in[13]};
  const float* lw1 = (const float*)d_in[14]; const float* lb1 = (const float*)d_in[15];
  const float* lw2 = (const float*)d_in[16]; const float* lb2 = (const float*)d_in[17];
  const float* lw3 = (const float*)d_in[18]; const float* lb3 = (const float*)d_in[19];
  const float* lw4 = (const float*)d_in[20]; const float* lb4 = (const float*)d_in[21];
  const float* lw5 = (const float*)d_in[22]; const float* lb5 = (const float*)d_in[23];

  const int N = in_sizes[0] / 16;     // 100000
  const int E = in_sizes[1] / 2;      // 1600000

  // Workspace layout (256B aligned blocks)
  char* ws = (char*)d_ws;
  size_t off = 0;
  auto carve = [&](size_t bytes) -> float* {
    float* p = (float*)(ws + off);
    off += (bytes + 255) & ~(size_t)255;
    return p;
  };
  float* deg  = carve((size_t)N * sizeof(float));              // [N]
  float* Tbuf = carve((size_t)N * HEADS * sizeof(float));      // [N,4]
  float* Ybuf = carve((size_t)N * 256 * sizeof(float));        // [N,<=256]
  float* agg  = carve((size_t)N * 64 * sizeof(float));         // [N,<=64]
  float* Hbuf = carve((size_t)N * 64 * sizeof(float));         // [N,<=64]
  (void)ws_size; (void)n_in; (void)out_size;

  // Degree (shared across layers)
  fill_f32<<<(N + 255) / 256, 256, 0, stream>>>(deg, N, 0.0f);
  degree_kernel<<<(E + 255) / 256, 256, 0, stream>>>(ei + E, E, deg);

  const int Kin[3] = {16, 16, 32};
  const int Fob[3] = {16, 32, 64};

  const float* cur = x;
  for (int l = 0; l < 3; ++l) {
    const int K = Kin[l];
    const int Fo = Fob[l];
    const int C = HEADS * Fo;

    fill_f32<<<((N * Fo) + 255) / 256, 256, 0, stream>>>(agg, N * Fo, 0.0f);
    proj_u_kernel<<<(N + 255) / 256, 256, 0, stream>>>(cur, Uw[l], Tbuf, N, K);

    dim3 grid((N + 127) / 128, C / 32);
    gemm_wmma_kernel<<<grid, 256, 0, stream>>>(cur, Ww[l], Ybuf, N, K, C);

    edge_kernel<<<(E + 7) / 8, 256, 0, stream>>>(ei, E, Tbuf, cv[l], Ybuf, agg, Fo);
    finalize_kernel<<<((N * Fo) + 255) / 256, 256, 0, stream>>>(agg, deg, bw[l], Hbuf, N, Fo);
    cur = Hbuf;
  }

  mlp_tail_kernel<<<(N + 255) / 256, 256, 0, stream>>>(Hbuf, lw1, lb1, lw2, lb2, lw3, lb3,
                                                       lw4, lb4, lw5, lb5, (float*)d_out, N);
}
